// CellBox_72000831750640
// MI455X (gfx1250) — matrix operand order repaired and checked by reading, not compiled, not run on hardware
//
#include <hip/hip_runtime.h>

#define NX 1024
#define NB 1024

typedef __attribute__((ext_vector_type(16))) _Float16 v16h;
typedef __attribute__((ext_vector_type(8)))  _Float16 v8h;
typedef __attribute__((ext_vector_type(8)))  float    v8f;

__device__ __forceinline__ float softplus_f(float x) {
    // numerically stable log1p(exp(x))
    return (x > 0.0f) ? (x + log1pf(expf(-x))) : log1pf(expf(x));
}

// ---------------------------------------------------------------------------
// Prep kernels
// ---------------------------------------------------------------------------
__global__ void prep_W(const float* __restrict__ W, const float* __restrict__ mask,
                       _Float16* __restrict__ Wh) {
    int i = blockIdx.x * blockDim.x + threadIdx.x;
    if (i < NX * NX) Wh[i] = (_Float16)(W[i] * mask[i]);
}

__global__ void prep_vec(const float* __restrict__ araw, const float* __restrict__ eraw,
                         float* __restrict__ alpha, float* __restrict__ eps) {
    int i = blockIdx.x * blockDim.x + threadIdx.x;
    if (i < NX) {
        alpha[i] = softplus_f(araw[i]);
        eps[i]   = softplus_f(eraw[i]);
    }
}

__global__ void prep_x0(const float* __restrict__ y0, float* __restrict__ S,
                        _Float16* __restrict__ T) {
    int i = blockIdx.x * blockDim.x + threadIdx.x;   // i = m*NB + b
    if (i < NX * NB) {
        int m = i / NB, b = i % NB;
        float v = y0[i];
        S[i] = v;
        T[(size_t)b * NX + m] = (_Float16)v;         // transposed f16 copy for WMMA B-operand
    }
}

// ---------------------------------------------------------------------------
// Fused GEMM + Heun epilogue. Wave tile: 64 (M) x 32 (N); block = 4 waves
// arranged 2x2 -> 128x64 block tile; grid (16, 8).
//   acc = Wm @ state  (f16 in, f32 WMMA accumulate), then per-element:
//   MODE 0: k1 = eps*tanh(acc+mu) - alpha*x;  outF=k1,  outT=f16(x + DT*k1)
//   MODE 1: y  = x + DT*k1; k2 = eps*tanh(acc+mu) - alpha*y;
//           xn = x + DT/2*(k1+k2);            outF=xn,  outT=f16(xn)
//   MODE 2: dx = eps*tanh(acc+mu) - alpha*x;  outF=dx   (no outT)
// ---------------------------------------------------------------------------
template <int MODE>
__global__ __launch_bounds__(128)
void fused_gemm(const _Float16* __restrict__ Wh,   // A: (NX x NX) row-major
                const _Float16* __restrict__ BT,   // B transposed: BT[b*NX + k]
                const float* __restrict__ x,       // f32 state (NX x NB)
                const float* __restrict__ k1,      // f32 k1 (MODE==1)
                const float* __restrict__ mu,      // (NB x NX)
                const float* __restrict__ alpha,
                const float* __restrict__ eps,
                float* __restrict__ outF,          // (NX x NB)
                _Float16* __restrict__ outT)       // (NB x NX) transposed f16
{
    const int lane = threadIdx.x & 31;
    const int wave = threadIdx.x >> 5;
    const int wm = wave >> 1, wn = wave & 1;
    const int m0 = blockIdx.y * 128 + wm * 64;     // 64 rows of C per wave
    const int b0 = blockIdx.x * 64 + wn * 32;      // 32 cols of C per wave
    const int l15 = lane & 15;
    const int hi  = lane >> 4;                     // 0: lanes 0-15, 1: lanes 16-31

    v8f acc[4][2] = {};

    // A row pointers for the four M-fragments of this wave
    const _Float16* Arow[4];
    Arow[0] = Wh + (size_t)(m0 + l15) * NX;
    Arow[1] = Arow[0] + (size_t)16 * NX;
    Arow[2] = Arow[0] + (size_t)32 * NX;
    Arow[3] = Arow[0] + (size_t)48 * NX;
    // B column pointers (transposed storage -> K contiguous per lane)
    const _Float16* Bcol0 = BT + (size_t)(b0 + l15) * NX;
    const _Float16* Bcol1 = Bcol0 + (size_t)16 * NX;

    for (int k0 = 0; k0 < NX; k0 += 32) {
        // A fragment (16x32; this code's contiguous k order, matched by B): lanes 0-15 hold K=k0..k0+15,
        // lanes 16-31 hold K=k0+16..k0+31.
        const int ka = k0 + hi * 16;
        v16h a[4];
#pragma unroll
        for (int mi = 0; mi < 4; ++mi) {
            a[mi] = *(const v16h*)(Arow[mi] + ka);
        }
        // B fragment (32x16): lanes 0-15 K=k0..k0+15, lanes 16-31 K=k0+16..k0+31;
        // column = b0 + (lane&15); contiguous 32B per lane.
        const int kb = k0 + hi * 16;
        v16h bf0 = *(const v16h*)(Bcol0 + kb);
        v16h bf1 = *(const v16h*)(Bcol1 + kb);

#pragma unroll
        for (int mi = 0; mi < 4; ++mi) {
            acc[mi][0] = __builtin_amdgcn_wmma_f32_16x16x32_f16(false, a[mi], false, bf0,
                                                                (short)0, acc[mi][0], false, false);
            acc[mi][1] = __builtin_amdgcn_wmma_f32_16x16x32_f16(false, a[mi], false, bf1,
                                                                (short)0, acc[mi][1], false, false);
        }
    }

    // Epilogue. C layout: lanes 0-15 -> N=lane, M=mf+v; lanes 16-31 -> N=lane-16, M=mf+8+v.
#pragma unroll
    for (int mi = 0; mi < 4; ++mi) {
        const int mbase = m0 + mi * 16 + hi * 8;
#pragma unroll
        for (int ni = 0; ni < 2; ++ni) {
            const int b = b0 + ni * 16 + l15;
            v8f A = acc[mi][ni];
            v8f muv = *(const v8f*)(mu + (size_t)b * NX + mbase);  // mu_t[m,b] = mu[b,m]
            v8h packT;
#pragma unroll
            for (int v = 0; v < 8; ++v) {
                const int m = mbase + v;
                const size_t idx = (size_t)m * NB + b;
                const float t = eps[m] * tanhf(A[v] + muv[v]);
                float res = 0.0f;
                if (MODE == 0) {
                    float xv  = x[idx];
                    float k1v = t - alpha[m] * xv;
                    outF[idx] = k1v;
                    res = xv + 0.1f * k1v;                 // y = x + DT*k1
                } else if (MODE == 1) {
                    float xv  = x[idx];
                    float k1v = k1[idx];
                    float yv  = xv + 0.1f * k1v;
                    float k2v = t - alpha[m] * yv;
                    float xn  = xv + 0.05f * (k1v + k2v);  // x + DT/2*(k1+k2)
                    outF[idx] = xn;
                    res = xn;
                } else {
                    float xv = x[idx];
                    outF[idx] = t - alpha[m] * xv;         // dx
                }
                packT[v] = (_Float16)res;
            }
            if (MODE != 2) {
                *(v8h*)(outT + (size_t)b * NX + mbase) = packT;  // 16B packed store
            }
        }
    }
}

// ---------------------------------------------------------------------------
// Final reduction: mean/sd over last two states + yhat transpose.
//   sd with ddof=1 over 2 samples == |a-b|/sqrt(2)
// ---------------------------------------------------------------------------
__global__ void finalize(const float* __restrict__ yf, const float* __restrict__ yp,
                         float* __restrict__ out) {
    int i = blockIdx.x * blockDim.x + threadIdx.x;   // i = m*NB + b
    if (i < NX * NB) {
        float a = yf[i], p = yp[i];
        out[i] = 0.5f * (a + p);                                        // mean
        out[(size_t)NX * NB + i] = fabsf(a - p) * 0.7071067811865476f;  // sd (ddof=1)
        int m = i / NB, b = i % NB;
        out[(size_t)3 * NX * NB + (size_t)b * NX + m] = a;              // yhat = y_final.T
    }
}

// ---------------------------------------------------------------------------
extern "C" void kernel_launch(void* const* d_in, const int* in_sizes, int n_in,
                              void* d_out, int out_size, void* d_ws, size_t ws_size,
                              hipStream_t stream) {
    const float* y0   = (const float*)d_in[0];
    const float* mu   = (const float*)d_in[1];
    const float* W    = (const float*)d_in[2];
    const float* araw = (const float*)d_in[3];
    const float* eraw = (const float*)d_in[4];
    const float* mask = (const float*)d_in[5];
    float* out = (float*)d_out;

    char* ws = (char*)d_ws;
    const size_t MB = 1u << 20;
    _Float16* Wh  = (_Float16*)(ws);               // 2 MB  masked W, f16
    float*    S0  = (float*)(ws + 2 * MB);         // 4 MB  state ping
    float*    S1  = (float*)(ws + 6 * MB);         // 4 MB  state pong
    float*    K1  = (float*)(ws + 10 * MB);        // 4 MB  k1
    _Float16* T0  = (_Float16*)(ws + 14 * MB);     // 2 MB  state ping, f16 transposed
    _Float16* T1  = (_Float16*)(ws + 16 * MB);     // 2 MB  state pong, f16 transposed
    _Float16* Ty  = (_Float16*)(ws + 18 * MB);     // 2 MB  y = x+DT*k1, f16 transposed
    float*    alp = (float*)(ws + 20 * MB);        // 4 KB
    float*    epv = (float*)(ws + 20 * MB + 4096); // 4 KB

    prep_W<<<(NX * NX + 255) / 256, 256, 0, stream>>>(W, mask, Wh);
    prep_vec<<<(NX + 255) / 256, 256, 0, stream>>>(araw, eraw, alp, epv);
    prep_x0<<<(NX * NB + 255) / 256, 256, 0, stream>>>(y0, S0, T0);

    float*    S[2] = {S0, S1};
    _Float16* T[2] = {T0, T1};
    dim3 grid(16, 8), block(128);   // 16 tiles of 64 in N, 8 tiles of 128 in M

    for (int t = 0; t < 100; ++t) {
        float*    xin  = S[t & 1];
        _Float16* tin  = T[t & 1];
        float*    xout = S[(t + 1) & 1];
        _Float16* tout = T[(t + 1) & 1];
        // k1 = dxdt(x); Ty = f16(x + DT*k1)
        fused_gemm<0><<<grid, block, 0, stream>>>(Wh, tin, xin, nullptr, mu, alp, epv, K1, Ty);
        // k2 = dxdt(y); xn = x + DT/2*(k1+k2)
        fused_gemm<1><<<grid, block, 0, stream>>>(Wh, Ty, xin, K1, mu, alp, epv, xout, tout);
    }
    // After 100 steps: ys[99] -> S0/T0, ys[98] -> S1.
    // dx = dxdt(y_final) -> rows [2*NX, 3*NX) of convergence_metric
    fused_gemm<2><<<grid, block, 0, stream>>>(Wh, T0, S0, nullptr, mu, alp, epv,
                                              out + (size_t)2 * NX * NB, nullptr);
    finalize<<<(NX * NB + 255) / 256, 256, 0, stream>>>(S0, S1, out);
}
